// QuantumReuploadingHead_50929722196751
// MI455X (gfx1250) — matrix operand order repaired and checked by reading, not compiled, not run on hardware
//
#include <hip/hip_runtime.h>
#include <hip/hip_bf16.h>
#include <math.h>

// ---------------------------------------------------------------------------
// Quantum re-uploading head for MI455X (gfx1250, wave32).
//  Kernel 1: one wave32 per batch element; 1024-amplitude statevector held
//            entirely in VGPRs (32 complex per lane). Gates on wires 5..9 are
//            in-lane register ops; wires 0..4 use ds_swizzle lane exchanges
//            (immediate XOR mask -> no address setup, unlike ds_bpermute).
//            CNOT ring is register permutation / conditional select / swizzle.
//            Emits 20 measurement features per batch element into d_ws.
//  Kernel 2: readout MLP feats[4096,20] @ W1^T -> silu -> @ W2^T using
//            v_wmma_f32_16x16x32_f16 (K padded 20->32), fused epilogue with
//            hardware v_rcp_f32 sigmoid (precision irrelevant at this scale).
// ---------------------------------------------------------------------------

#define P_Q   10
#define LQ_N  4
#define B_N   4096
#define HID_N 32
#define NMEAS 20

typedef __attribute__((ext_vector_type(16))) _Float16 v16h;
typedef __attribute__((ext_vector_type(8)))  float    v8f;

// Lane exchange via ds_swizzle_b32 group-of-32 mode:
//   new_lane = ((lane & and_mask) | or_mask) ^ xor_mask
// offset = {1'b0, xor[14:10], or[9:5], and[4:0]}; immediate-encoded, no VGPR
// address setup (cheaper than the ds_bpermute path __shfl_xor lowers to).
template <int M>
__device__ __forceinline__ float lane_xor(float v) {
  constexpr int offset = (M << 10) | 0x1f;
  return __int_as_float(__builtin_amdgcn_ds_swizzle(__float_as_int(v), offset));
}

// complex 2x2 gate U stored as 8 floats: {00r,00i,01r,01i,10r,10i,11r,11i}
__device__ __forceinline__ void make_gate(int l, int k, float th,
                                          const float* __restrict__ rotw,
                                          const float* __restrict__ esc,
                                          const float* __restrict__ ebi,
                                          float U[8]) {
  const int idx2 = (l * P_Q + k) * 2;
  const float t0 = esc[idx2 + 0] * th + ebi[idx2 + 0];
  const float t1 = esc[idx2 + 1] * th + ebi[idx2 + 1];
  const float c1 = __cosf(0.5f * t1), s1 = __sinf(0.5f * t1);
  const float ec = __cosf(0.5f * t0), es = __sinf(0.5f * t0);
  // U_enc = RX(t1) @ RZ(t0)
  const float E00r =  c1 * ec, E00i = -c1 * es;
  const float E01r =  s1 * es, E01i = -s1 * ec;
  const float E10r = -s1 * es, E10i = -s1 * ec;
  const float E11r =  c1 * ec, E11i =  c1 * es;
  // U_rot = RZ(w2) RY(w1) RZ(w0)
  const int idx3 = (l * P_Q + k) * 3;
  const float w0 = rotw[idx3 + 0], w1 = rotw[idx3 + 1], w2 = rotw[idx3 + 2];
  const float ct = __cosf(0.5f * w1), st = __sinf(0.5f * w1);
  const float A  = 0.5f * (w0 + w2), Bq = 0.5f * (w0 - w2);
  const float cA = __cosf(A),  sA = __sinf(A);
  const float cB = __cosf(Bq), sB = __sinf(Bq);
  const float R00r =  ct * cA, R00i = -ct * sA;
  const float R01r = -st * cB, R01i = -st * sB;
  const float R10r =  st * cB, R10i = -st * sB;
  const float R11r =  ct * cA, R11i =  ct * sA;
  // U = R @ E (complex)
  U[0] = R00r*E00r - R00i*E00i + R01r*E10r - R01i*E10i;
  U[1] = R00r*E00i + R00i*E00r + R01r*E10i + R01i*E10r;
  U[2] = R00r*E01r - R00i*E01i + R01r*E11r - R01i*E11i;
  U[3] = R00r*E01i + R00i*E01r + R01r*E11i + R01i*E11r;
  U[4] = R10r*E00r - R10i*E00i + R11r*E10r - R11i*E10i;
  U[5] = R10r*E00i + R10i*E00r + R11r*E10i + R11i*E10r;
  U[6] = R10r*E01r - R10i*E01i + R11r*E11r - R11i*E11i;
  U[7] = R10r*E01i + R10i*E01r + R11r*E11i + R11i*E11r;
}

// Apply single-qubit gate on wire K. Global index = (lane<<5)|j, wire0 = MSB.
// K>=5 -> local bit (9-K); K<=4 -> lane bit (4-K).
template <int K>
__device__ __forceinline__ void apply_gate(float (&sr)[32], float (&si)[32],
                                           int lane, const float U[8]) {
  const float u00r = U[0], u00i = U[1], u01r = U[2], u01i = U[3];
  const float u10r = U[4], u10i = U[5], u11r = U[6], u11i = U[7];
  if constexpr (K >= 5) {
    constexpr int M = 1 << (9 - K);
#pragma unroll
    for (int j = 0; j < 32; ++j) {
      if ((j & M) == 0) {
        const int j1 = j | M;
        const float ar = sr[j],  ai = si[j];
        const float br = sr[j1], bi = si[j1];
        sr[j]  = u00r*ar - u00i*ai + u01r*br - u01i*bi;
        si[j]  = u00r*ai + u00i*ar + u01r*bi + u01i*br;
        sr[j1] = u10r*ar - u10i*ai + u11r*br - u11i*bi;
        si[j1] = u10r*ai + u10i*ar + u11r*bi + u11i*br;
      }
    }
  } else {
    constexpr int LM = 1 << (4 - K);
    const bool hi = (lane & LM) != 0;
    // my coefficient / partner coefficient (complex)
    const float cmr = hi ? u11r : u00r, cmi = hi ? u11i : u00i;
    const float cor = hi ? u10r : u01r, coi = hi ? u10i : u01i;
#pragma unroll
    for (int j = 0; j < 32; ++j) {
      const float pr = lane_xor<LM>(sr[j]);
      const float pi = lane_xor<LM>(si[j]);
      const float mr = sr[j], mi = si[j];
      sr[j] = cmr*mr - cmi*mi + cor*pr - coi*pi;
      si[j] = cmr*mi + cmi*mr + cor*pi + coi*pr;
    }
  }
}

// CNOT ring: CNOT(c, (c+1)%10) for c = 0..9
__device__ __forceinline__ void cnot_ring(float (&sr)[32], float (&si)[32],
                                          int lane) {
  // c = 0..3 : control = lane bit (4-c), target = lane bit (3-c)
#pragma unroll
  for (int c = 0; c < 4; ++c) {
    constexpr int CMt[4] = {16, 8, 4, 2};
    constexpr int TMt[4] = {8, 4, 2, 1};
    const bool ctl = (lane & CMt[c]) != 0;
#pragma unroll
    for (int j = 0; j < 32; ++j) {
      float pr, pi;
      if (c == 0)      { pr = lane_xor<8>(sr[j]); pi = lane_xor<8>(si[j]); }
      else if (c == 1) { pr = lane_xor<4>(sr[j]); pi = lane_xor<4>(si[j]); }
      else if (c == 2) { pr = lane_xor<2>(sr[j]); pi = lane_xor<2>(si[j]); }
      else             { pr = lane_xor<1>(sr[j]); pi = lane_xor<1>(si[j]); }
      sr[j] = ctl ? pr : sr[j];
      si[j] = ctl ? pi : si[j];
    }
  }
  // c = 4 : control = lane bit 0, target = local bit 4 (swap j <-> j^16)
  {
    const bool ctl = (lane & 1) != 0;
#pragma unroll
    for (int j = 0; j < 16; ++j) {
      const int j1 = j | 16;
      const float ar = sr[j],  ai = si[j];
      const float br = sr[j1], bi = si[j1];
      sr[j]  = ctl ? br : ar;  si[j]  = ctl ? bi : ai;
      sr[j1] = ctl ? ar : br;  si[j1] = ctl ? ai : bi;
    }
  }
  // c = 5..8 : pure in-lane permutation (register renaming)
#pragma unroll
  for (int c = 5; c < 9; ++c) {
    const int CMl = 1 << (9 - c);
    const int TMl = 1 << (8 - c);
#pragma unroll
    for (int j = 0; j < 32; ++j) {
      if ((j & CMl) != 0 && (j & TMl) == 0) {
        const int j1 = j | TMl;
        float t;
        t = sr[j]; sr[j] = sr[j1]; sr[j1] = t;
        t = si[j]; si[j] = si[j1]; si[j1] = t;
      }
    }
  }
  // c = 9 : control = local bit 0 (compile-time!), target = lane bit 4
#pragma unroll
  for (int j = 1; j < 32; j += 2) {
    sr[j] = lane_xor<16>(sr[j]);
    si[j] = lane_xor<16>(si[j]);
  }
}

__global__ __launch_bounds__(256) void qc_circuit_kernel(
    const float* __restrict__ theta,   // [B, 10]
    const float* __restrict__ rotw,    // [4,10,3]
    const float* __restrict__ esc,     // [4,10,2]
    const float* __restrict__ ebi,     // [4,10,2]
    float* __restrict__ feats)         // [B, 20]
{
  const int lane = threadIdx.x & 31;
  const int wave = threadIdx.x >> 5;
  const int b    = blockIdx.x * 8 + wave;

  float thv[P_Q];
#pragma unroll
  for (int k = 0; k < P_Q; ++k) thv[k] = theta[b * P_Q + k];

  // |+...+> after Hadamards: uniform real amplitude 1/sqrt(1024)
  float sr[32], si[32];
#pragma unroll
  for (int j = 0; j < 32; ++j) { sr[j] = 0.03125f; si[j] = 0.0f; }

  for (int l = 0; l < LQ_N; ++l) {
    float U[8];
#define QC_GATE(KK) make_gate(l, KK, thv[KK], rotw, esc, ebi, U); \
                    apply_gate<KK>(sr, si, lane, U);
    QC_GATE(0) QC_GATE(1) QC_GATE(2) QC_GATE(3) QC_GATE(4)
    QC_GATE(5) QC_GATE(6) QC_GATE(7) QC_GATE(8) QC_GATE(9)
#undef QC_GATE
    cnot_ring(sr, si, lane);
  }

  // ---- measurement: Z_k (k=0..9) and Z_k Z_{k+1 mod 10} ----
  // lane-bit wires (0..4): sign depends only on lane -> share Ptot.
  // local-bit wires (5..9): compile-time signed accumulators.
  float Ptot = 0.f;
  float A5 = 0.f, A6 = 0.f, A7 = 0.f, A8 = 0.f, A9 = 0.f;
  float B5 = 0.f, B6 = 0.f, B7 = 0.f, B8 = 0.f;
#pragma unroll
  for (int j = 0; j < 32; ++j) {
    const float pj = sr[j]*sr[j] + si[j]*si[j];
    Ptot += pj;
    const bool b4 = (j >> 4) & 1, b3 = (j >> 3) & 1, b2 = (j >> 2) & 1;
    const bool b1 = (j >> 1) & 1, b0 = j & 1;
    A5 += b4 ? -pj : pj;              // wire 5 (bit 4)
    A6 += b3 ? -pj : pj;              // wire 6
    A7 += b2 ? -pj : pj;              // wire 7
    A8 += b1 ? -pj : pj;              // wire 8
    A9 += b0 ? -pj : pj;              // wire 9
    B5 += (b4 ^ b3) ? -pj : pj;       // Z5 Z6
    B6 += (b3 ^ b2) ? -pj : pj;       // Z6 Z7
    B7 += (b2 ^ b1) ? -pj : pj;       // Z7 Z8
    B8 += (b1 ^ b0) ? -pj : pj;       // Z8 Z9
  }
  const float ls0 = (lane & 16) ? -1.f : 1.f;   // wire 0 -> lane bit 4
  const float ls1 = (lane &  8) ? -1.f : 1.f;
  const float ls2 = (lane &  4) ? -1.f : 1.f;
  const float ls3 = (lane &  2) ? -1.f : 1.f;
  const float ls4 = (lane &  1) ? -1.f : 1.f;   // wire 4 -> lane bit 0

  float f[NMEAS];
  f[0]  = ls0 * Ptot;  f[1] = ls1 * Ptot;  f[2] = ls2 * Ptot;
  f[3]  = ls3 * Ptot;  f[4] = ls4 * Ptot;
  f[5]  = A5;  f[6] = A6;  f[7] = A7;  f[8] = A8;  f[9] = A9;
  f[10] = ls0 * ls1 * Ptot;  f[11] = ls1 * ls2 * Ptot;
  f[12] = ls2 * ls3 * Ptot;  f[13] = ls3 * ls4 * Ptot;
  f[14] = ls4 * A5;          // Z4 Z5
  f[15] = B5;  f[16] = B6;  f[17] = B7;  f[18] = B8;
  f[19] = ls0 * A9;          // Z9 Z0

#pragma unroll
  for (int m = 0; m < NMEAS; ++m) {
    float v = f[m];
    v += lane_xor<1>(v);
    v += lane_xor<2>(v);
    v += lane_xor<4>(v);
    v += lane_xor<8>(v);
    v += lane_xor<16>(v);
    f[m] = v;
  }
  if (lane == 0) {
#pragma unroll
    for (int m = 0; m < NMEAS; ++m) feats[b * NMEAS + m] = f[m];
  }
}

// ---------------------------------------------------------------------------
// MLP readout with WMMA: h = silu(feats @ W1^T + b1); out = h @ W2^T + b2
// One wave per 16-row batch tile; two WMMAs (HID = 2 x 16 columns), K=20
// zero-padded to 32. A/B packed per CDNA5 16-bit WMMA VGPR layouts.
// ---------------------------------------------------------------------------
__device__ __forceinline__ float fast_silu(float x) {
  // x * sigmoid(x) with hardware v_exp/v_rcp (precision irrelevant here)
  return x * __builtin_amdgcn_rcpf(1.f + __expf(-x));
}

__global__ __launch_bounds__(256) void qc_mlp_kernel(
    const float* __restrict__ feats,   // [B, 20]
    const float* __restrict__ W1,      // [32, 20]
    const float* __restrict__ b1,      // [32]
    const float* __restrict__ W2,      // [1, 32]
    const float* __restrict__ b2,      // [1]
    float* __restrict__ out)           // [B]
{
  const int lane = threadIdx.x & 31;
  const int wave = threadIdx.x >> 5;
  const int tile = blockIdx.x * 8 + wave;        // 256 tiles of 16 rows
  const int m0   = tile * 16;
  const int half = lane >> 4;                    // 0: lanes 0-15, 1: 16-31
  const int ln   = lane & 15;

  // A (16x32 f16): lane row M=ln; element e -> K = e + 8*half + (e>=8 ? 8 : 0)
  v16h a;
#pragma unroll
  for (int e = 0; e < 16; ++e) {
    const int k = e + 8 * half + (e >= 8 ? 8 : 0);
    const float v = (k < NMEAS) ? feats[(m0 + ln) * NMEAS + k] : 0.f;
    a[e] = (_Float16)v;
  }
  // B (32x16 f16): lane col N=ln; element e -> K = e + 8*half + (e>=8 ? 8 : 0)
  v16h bA, bB;
#pragma unroll
  for (int e = 0; e < 16; ++e) {
    const int k = e + 8 * half + (e >= 8 ? 8 : 0);
    const float v0 = (k < NMEAS) ? W1[(ln)      * NMEAS + k] : 0.f;
    const float v1 = (k < NMEAS) ? W1[(16 + ln) * NMEAS + k] : 0.f;
    bA[e] = (_Float16)v0;
    bB[e] = (_Float16)v1;
  }

  v8f c0 = {}, c1 = {};
  c0 = __builtin_amdgcn_wmma_f32_16x16x32_f16(false, a, false, bA,
                                              (short)0, c0, false, false);
  c1 = __builtin_amdgcn_wmma_f32_16x16x32_f16(false, a, false, bB,
                                              (short)0, c1, false, false);

  const float bias0 = b1[ln], bias1 = b1[16 + ln];
  const float w2a = W2[ln], w2b = W2[16 + ln];
  const float bb2 = b2[0];

  // C layout: VGPR r, lanes 0-15 -> (M=r, N=ln); lanes 16-31 -> (M=r+8, N=ln)
#pragma unroll
  for (int r = 0; r < 8; ++r) {
    const float h0 = fast_silu(c0[r] + bias0);
    const float h1 = fast_silu(c1[r] + bias1);
    float t = h0 * w2a + h1 * w2b;
    // reduce over hidden (16 lanes within this half-wave)
    t += lane_xor<1>(t);
    t += lane_xor<2>(t);
    t += lane_xor<4>(t);
    t += lane_xor<8>(t);
    if (ln == 0) out[m0 + r + 8 * half] = t + bb2;
  }
}

extern "C" void kernel_launch(void* const* d_in, const int* in_sizes, int n_in,
                              void* d_out, int out_size, void* d_ws, size_t ws_size,
                              hipStream_t stream) {
  const float* theta = (const float*)d_in[0];
  const float* rotw  = (const float*)d_in[1];
  const float* esc   = (const float*)d_in[2];
  const float* ebi   = (const float*)d_in[3];
  const float* W1    = (const float*)d_in[4];
  const float* b1    = (const float*)d_in[5];
  const float* W2    = (const float*)d_in[6];
  const float* b2    = (const float*)d_in[7];
  float* out   = (float*)d_out;
  float* feats = (float*)d_ws;                   // B*20 floats = 320 KB scratch

  qc_circuit_kernel<<<B_N / 8, 256, 0, stream>>>(theta, rotw, esc, ebi, feats);
  qc_mlp_kernel<<<(B_N / 16) / 8, 256, 0, stream>>>(feats, W1, b1, W2, b2, out);
}
